// TransformerBlock_35768487641418
// MI455X (gfx1250) — compile-verified, matrix operand order checked by reading, not run on hardware
//
#include <hip/hip_runtime.h>
#include <math.h>

#define B_  64
#define T_  256
#define E_  1024
#define H_  16
#define HD_ 64
#define BT_ (B_ * T_)
#define NEG_BIG (-1.0e30f)

#define USE_TDM 1   // Tensor Data Mover path for GEMM tile staging

typedef __attribute__((ext_vector_type(16))) __bf16 v16bf;
typedef __attribute__((ext_vector_type(8)))  float  v8f;
typedef __attribute__((ext_vector_type(4)))  unsigned int v4u;
typedef __attribute__((ext_vector_type(8)))  int v8i;
typedef __attribute__((ext_vector_type(4)))  int v4i;

union FragC { v8f v; float f[8]; };

// 16-bit A/B fragment (16x32): dword j of the per-lane 8-dword fragment holds
// K-pair index ((j&4)<<1) + 4*hi + (j&3)   (hi = lane>=16)
__device__ __forceinline__ int kpair_idx(int j, int hi) {
  return ((j & 4) << 1) + (hi << 2) + (j & 3);
}

__device__ __forceinline__ v16bf load_frag(const unsigned int* row32, int hi) {
  union { v16bf v; unsigned int u[8]; } f;
#pragma unroll
  for (int j = 0; j < 8; ++j) f.u[j] = row32[kpair_idx(j, hi)];
  return f.v;
}

#if USE_TDM
// LDS byte offset of a __shared__ object: generic -> AS(3) cast, then ptrtoint
// (AS3 pointers are LDS-base-relative byte offsets).
__device__ __forceinline__ unsigned lds_offset(const void* p) {
  return (unsigned)(unsigned long long)(__attribute__((address_space(3))) const char*)p;
}

// One 2D TDM descriptor: tile = 32 bf16 (contiguous, 64B rows) x 16 rows,
// row stride = strideElems, LDS rows padded 64B -> 80B (pad_interval=16dw,
// pad_amount=4dw) to match LDS_STR = 40 bf16.
__device__ __forceinline__ void tdm_load_tile(unsigned ldsByteAddr,
                                              const __bf16* gsrc,
                                              unsigned strideElems) {
  unsigned long long ga = (unsigned long long)(const void*)gsrc;
  const unsigned td0 = 1u << 20, td1 = 1u << 20;    // large dims: no OOB clip
  v4u g0;
  g0[0] = 1u;                                       // count=1, user descriptor
  g0[1] = ldsByteAddr;                              // bits 63:32  lds_addr
  g0[2] = (unsigned)(ga & 0xFFFFFFFFu);             // bits 95:64  global_addr lo
  g0[3] = (unsigned)((ga >> 32) & 0x01FFFFFFu)      // bits 120:96 global_addr hi
        | 0x80000000u;                              // bits 127:126 type=2
  v8i g1;
  g1[0] = (int)((1u << 16) |                        // data_size = 2 bytes
                (1u << 20) |                        // pad_enable
                (3u << 22) |                        // pad_interval: 16 dwords
                (3u << 25));                        // pad_amount: 4 dwords
  g1[1] = (int)((td0 & 0xFFFFu) << 16);             // tensor_dim0[15:0]
  g1[2] = (int)(((td0 >> 16) & 0xFFFFu) | ((td1 & 0xFFFFu) << 16));
  g1[3] = (int)(((td1 >> 16) & 0xFFFFu) | (32u << 16)); // tile_dim0 = 32
  g1[4] = 16;                                       // tile_dim1 = 16, tile_dim2 = 0
  g1[5] = (int)strideElems;                         // tensor_dim0_stride lo
  g1[6] = 0;                                        // stride hi / dim1_stride lo
  g1[7] = 0;
  v4i z4 = {0, 0, 0, 0};
  v8i z8 = {0, 0, 0, 0, 0, 0, 0, 0};
  // 6-arg form (amdgpu-toolchain / clang-23): groups 2,3 unused (2D tensor),
  // 5th arg is the extra descriptor group, last arg is cache policy.
  __builtin_amdgcn_tensor_load_to_lds(g0, g1, z4, z4, z8, 0);
}
#endif

// ---------------------------------------------------------------------------
// Weight conversion: emit W^T (row-major [N][K], bf16) so GEMM B-tiles are
// plain row-major slices (B-fragment lane n == row n of W^T).
// ---------------------------------------------------------------------------
__global__ void cvt_headmajor_T_kernel(const float* __restrict__ w, __bf16* __restrict__ out) {
  // w: (H, E, HD); out: WT[c][e], c = h*64+d
  int idx = blockIdx.x * 256 + threadIdx.x;   // over E*E
  int c = idx >> 10;
  int e = idx & 1023;
  int h = c >> 6;
  int d = c & 63;
  out[idx] = (__bf16)w[((size_t)h * E_ + e) * HD_ + d];
}

__global__ void cvt_plain_T_kernel(const float* __restrict__ w, __bf16* __restrict__ out) {
  // w: [K][N]; out: WT[n][k]
  int idx = blockIdx.x * 256 + threadIdx.x;
  int n = idx >> 10;
  int k = idx & 1023;
  out[idx] = (__bf16)w[(size_t)k * E_ + n];
}

// ---------------------------------------------------------------------------
// LayerNorm (one block per row of E=1024), output bf16
// ---------------------------------------------------------------------------
__launch_bounds__(256)
__global__ void ln_kernel(const float* __restrict__ x, const float* __restrict__ g,
                          const float* __restrict__ b, __bf16* __restrict__ out) {
  __shared__ float sbuf[256];
  const int tid = threadIdx.x;
  const int row = blockIdx.x;
  const float* xr = x + (size_t)row * E_;
  float v0[4];
  float s = 0.0f;
#pragma unroll
  for (int i = 0; i < 4; ++i) { v0[i] = xr[tid + i * 256]; s += v0[i]; }
  sbuf[tid] = s; __syncthreads();
  for (int off = 128; off > 0; off >>= 1) {
    if (tid < off) sbuf[tid] += sbuf[tid + off];
    __syncthreads();
  }
  float mu = sbuf[0] * (1.0f / E_);
  __syncthreads();
  float q = 0.0f;
#pragma unroll
  for (int i = 0; i < 4; ++i) { float d = v0[i] - mu; q += d * d; }
  sbuf[tid] = q; __syncthreads();
  for (int off = 128; off > 0; off >>= 1) {
    if (tid < off) sbuf[tid] += sbuf[tid + off];
    __syncthreads();
  }
  float rstd = rsqrtf(sbuf[0] * (1.0f / E_) + 1e-5f);
#pragma unroll
  for (int i = 0; i < 4; ++i) {
    int e = tid + i * 256;
    out[(size_t)row * E_ + e] = (__bf16)((v0[i] - mu) * rstd * g[e] + b[e]);
  }
}

// ---------------------------------------------------------------------------
// WMMA GEMM: C[M,N] = A[M,K](bf16) @ W[K,N], W given transposed (WT[N][K]).
// Fixed M=16384, N=K=1024. 128x128x32 block, 8 waves (2x4), wave = 64x32.
// TDM double-buffered LDS staging; epilogue fully compile-time specialized.
// ---------------------------------------------------------------------------
#define BM 128
#define BN 128
#define BK 32
#define NSTEPS (E_ / BK)
#define LDS_STR 40   // padded bf16 row stride (64B row + 16B pad)

template <bool HAS_BIAS, bool HAS_RES, bool RELU, bool OUT_F, bool OUT_B, bool OUT_VT>
__launch_bounds__(256)
__global__ void gemm_bf16_kernel(const __bf16* __restrict__ A, const __bf16* __restrict__ WT,
                                 const float* __restrict__ bias, const float* __restrict__ resid,
                                 float* __restrict__ outF, __bf16* __restrict__ outB,
                                 __bf16* __restrict__ outVT) {
  __shared__ __bf16 sA[2][BM * LDS_STR];
  __shared__ __bf16 sB[2][BN * LDS_STR];
  const int tid  = threadIdx.x;
  const int lane = tid & 31, wave = tid >> 5;
  const int lm = lane & 15, hi = lane >> 4;
  const int wm = wave >> 2, wn = wave & 3;
  const int blockM = blockIdx.y * BM, blockN = blockIdx.x * BN;

  FragC acc[4][2];
#pragma unroll
  for (int i = 0; i < 4; ++i)
#pragma unroll
    for (int j = 0; j < 2; ++j)
#pragma unroll
      for (int r = 0; r < 8; ++r) acc[i][j].f[r] = 0.0f;

#if USE_TDM
  // Each wave DMAs its own 16-row slice of the A tile and of the B tile.
  const __bf16* gA = A  + (size_t)(blockM + wave * 16) * E_;
  const __bf16* gB = WT + (size_t)(blockN + wave * 16) * E_;
  const unsigned ldsA0 = lds_offset(&sA[0][0]) + wave * 16 * LDS_STR * 2;
  const unsigned ldsB0 = lds_offset(&sB[0][0]) + wave * 16 * LDS_STR * 2;
  const unsigned bufStride = (unsigned)(BM * LDS_STR * 2);

  tdm_load_tile(ldsA0, gA, E_);
  tdm_load_tile(ldsB0, gB, E_);
  __builtin_amdgcn_s_wait_tensorcnt(0);
  __syncthreads();

  for (int kt = 0; kt < NSTEPS; ++kt) {
    const int cur = kt & 1;
    if (kt + 1 < NSTEPS) {
      const int k1 = (kt + 1) * BK;
      tdm_load_tile(ldsA0 + (cur ^ 1) * bufStride, gA + k1, E_);
      tdm_load_tile(ldsB0 + (cur ^ 1) * bufStride, gB + k1, E_);
    }
    v16bf aF[4], bF[2];
#pragma unroll
    for (int mt = 0; mt < 4; ++mt)
      aF[mt] = load_frag((const unsigned int*)(&sA[cur][0] + (wm * 64 + mt * 16 + lm) * LDS_STR), hi);
#pragma unroll
    for (int nt = 0; nt < 2; ++nt)
      bF[nt] = load_frag((const unsigned int*)(&sB[cur][0] + (wn * 32 + nt * 16 + lm) * LDS_STR), hi);
#pragma unroll
    for (int mt = 0; mt < 4; ++mt)
#pragma unroll
      for (int nt = 0; nt < 2; ++nt)
        acc[mt][nt].v = __builtin_amdgcn_wmma_f32_16x16x32_bf16(
            false, aF[mt], false, bF[nt], (short)0, acc[mt][nt].v, false, false);
    __builtin_amdgcn_s_wait_tensorcnt(0);
    __syncthreads();
  }
#else
  for (int kt = 0; kt < NSTEPS; ++kt) {
    const int k0 = kt * BK;
#pragma unroll
    for (int s = 0; s < 2; ++s) {
      int seg = tid + s * 256;
      int row = seg >> 2;
      int cs  = (seg & 3) * 8;
      uint4 va = *(const uint4*)(A + (size_t)(blockM + row) * E_ + k0 + cs);
      *(uint4*)(&sA[0][row * LDS_STR + cs]) = va;
      uint4 vb = *(const uint4*)(WT + (size_t)(blockN + row) * E_ + k0 + cs);
      *(uint4*)(&sB[0][row * LDS_STR + cs]) = vb;
    }
    __syncthreads();
    v16bf aF[4], bF[2];
#pragma unroll
    for (int mt = 0; mt < 4; ++mt)
      aF[mt] = load_frag((const unsigned int*)(&sA[0][0] + (wm * 64 + mt * 16 + lm) * LDS_STR), hi);
#pragma unroll
    for (int nt = 0; nt < 2; ++nt)
      bF[nt] = load_frag((const unsigned int*)(&sB[0][0] + (wn * 32 + nt * 16 + lm) * LDS_STR), hi);
#pragma unroll
    for (int mt = 0; mt < 4; ++mt)
#pragma unroll
      for (int nt = 0; nt < 2; ++nt)
        acc[mt][nt].v = __builtin_amdgcn_wmma_f32_16x16x32_bf16(
            false, aF[mt], false, bF[nt], (short)0, acc[mt][nt].v, false, false);
    __syncthreads();
  }
#endif

  // epilogue (all paths compile-time selected)
#pragma unroll
  for (int mt = 0; mt < 4; ++mt) {
#pragma unroll
    for (int nt = 0; nt < 2; ++nt) {
      int col = blockN + wn * 32 + nt * 16 + lm;
      float bv = 0.0f;
      if constexpr (HAS_BIAS) bv = bias[col];
#pragma unroll
      for (int r = 0; r < 8; ++r) {
        int row = blockM + wm * 64 + mt * 16 + r + hi * 8;
        float val = acc[mt][nt].f[r] + bv;
        if constexpr (HAS_RES) val += resid[(size_t)row * E_ + col];
        if constexpr (RELU)    val = val > 0.0f ? val : 0.0f;
        if constexpr (OUT_F)   outF[(size_t)row * E_ + col] = val;
        if constexpr (OUT_B)   outB[(size_t)row * E_ + col] = (__bf16)val;
        if constexpr (OUT_VT) {                  // [b][h][d][s] for attention V
          int bb = row >> 8, ss = row & 255;     // T_ = 256
          int hh = col >> 6, dd = col & 63;      // HD_ = 64
          outVT[(((size_t)bb * H_ + hh) * HD_ + dd) * T_ + ss] = (__bf16)val;
        }
      }
    }
  }
}

// ---------------------------------------------------------------------------
// Flash-style causal attention. scores[t,s] = k[t]·q[s]/8 (K·Q^T as in ref).
// One wave = 16 rows (t). Loop over 32-wide s chunks up to the diagonal.
// ---------------------------------------------------------------------------
#define PSTR 36

__launch_bounds__(128)
__global__ void attn_kernel(const __bf16* __restrict__ q, const __bf16* __restrict__ k,
                            const __bf16* __restrict__ vT, __bf16* __restrict__ o) {
  __shared__ __bf16 ldsP[4 * 16 * PSTR];
  const int tid  = threadIdx.x;
  const int lane = tid & 31, wave = tid >> 5;
  const int lm = lane & 15, hi = lane >> 4;
  const int bh = blockIdx.y;
  const int b = bh >> 4;                 // / H_
  const int h = bh & 15;
  const int t0 = blockIdx.x * 64 + wave * 16;
  __bf16* pbase = ldsP + wave * 16 * PSTR;

  v16bf ka[2];
  {
    const unsigned int* krow =
        (const unsigned int*)(k + ((size_t)b * T_ + t0 + lm) * E_ + h * HD_);
#pragma unroll
    for (int c = 0; c < 2; ++c) ka[c] = load_frag(krow + c * 16, hi);
  }

  FragC oacc[4];
#pragma unroll
  for (int n = 0; n < 4; ++n)
#pragma unroll
    for (int r = 0; r < 8; ++r) oacc[n].f[r] = 0.0f;

  float mrow[8], lrow[8];
#pragma unroll
  for (int r = 0; r < 8; ++r) { mrow[r] = NEG_BIG; lrow[r] = 0.0f; }

  const int lastChunk = (t0 + 15) >> 5;
  for (int sc = 0; sc <= lastChunk; ++sc) {
    const int sbase = sc * 32;

    FragC st[2];
#pragma unroll
    for (int ti = 0; ti < 2; ++ti) {
      int scol = sbase + ti * 16 + lm;
      const unsigned int* qrow =
          (const unsigned int*)(q + ((size_t)b * T_ + scol) * E_ + h * HD_);
      v16bf qb0 = load_frag(qrow, hi);
      v16bf qb1 = load_frag(qrow + 16, hi);
      FragC s;
#pragma unroll
      for (int r = 0; r < 8; ++r) s.f[r] = 0.0f;
      s.v = __builtin_amdgcn_wmma_f32_16x16x32_bf16(false, ka[0], false, qb0,
                                                    (short)0, s.v, false, false);
      s.v = __builtin_amdgcn_wmma_f32_16x16x32_bf16(false, ka[1], false, qb1,
                                                    (short)0, s.v, false, false);
      st[ti] = s;
    }

#pragma unroll
    for (int r = 0; r < 8; ++r) {
      int t = t0 + r + hi * 8;
      float p0 = st[0].f[r] * 0.125f;
      float p1 = st[1].f[r] * 0.125f;
      if (sbase + lm      > t) p0 = NEG_BIG;
      if (sbase + 16 + lm > t) p1 = NEG_BIG;
      float mx = fmaxf(p0, p1);
#pragma unroll
      for (int xd = 1; xd < 16; xd <<= 1) mx = fmaxf(mx, __shfl_xor(mx, xd, 32));
      float newm  = fmaxf(mrow[r], mx);
      float scale = __expf(mrow[r] - newm);
      float e0 = __expf(p0 - newm);
      float e1 = __expf(p1 - newm);
      float sum = e0 + e1;
#pragma unroll
      for (int xd = 1; xd < 16; xd <<= 1) sum += __shfl_xor(sum, xd, 32);
      lrow[r] = lrow[r] * scale + sum;
      mrow[r] = newm;
#pragma unroll
      for (int n = 0; n < 4; ++n) oacc[n].f[r] *= scale;
      pbase[(r + hi * 8) * PSTR + lm]      = (__bf16)e0;
      pbase[(r + hi * 8) * PSTR + 16 + lm] = (__bf16)e1;
    }
    asm volatile("" ::: "memory");  // same-wave LDS is in-order; block reordering

    v16bf pf = load_frag((const unsigned int*)(pbase + lm * PSTR), hi);

#pragma unroll
    for (int n = 0; n < 4; ++n) {
      int d = n * 16 + lm;
      const unsigned int* vrow =
          (const unsigned int*)(vT + (((size_t)b * H_ + h) * HD_ + d) * T_ + sbase);
      v16bf vb = load_frag(vrow, hi);
      oacc[n].v = __builtin_amdgcn_wmma_f32_16x16x32_bf16(false, pf, false, vb,
                                                          (short)0, oacc[n].v, false, false);
    }
  }

#pragma unroll
  for (int n = 0; n < 4; ++n) {
#pragma unroll
    for (int r = 0; r < 8; ++r) {
      int t = t0 + r + hi * 8;
      float val = oacc[n].f[r] / lrow[r];
      o[((size_t)b * T_ + t) * E_ + h * HD_ + n * 16 + lm] = (__bf16)val;
    }
  }
}

// ---------------------------------------------------------------------------
// Host launcher
// ---------------------------------------------------------------------------
extern "C" void kernel_launch(void* const* d_in, const int* in_sizes, int n_in,
                              void* d_out, int out_size, void* d_ws, size_t ws_size,
                              hipStream_t stream) {
  (void)in_sizes; (void)n_in; (void)out_size; (void)ws_size;
  const float* x      = (const float*)d_in[0];
  const float* ln1_g  = (const float*)d_in[1];
  const float* ln1_b  = (const float*)d_in[2];
  const float* wk     = (const float*)d_in[3];
  const float* wq     = (const float*)d_in[4];
  const float* wv     = (const float*)d_in[5];
  const float* w_proj = (const float*)d_in[6];
  const float* b_proj = (const float*)d_in[7];
  const float* ln2_g  = (const float*)d_in[8];
  const float* ln2_b  = (const float*)d_in[9];
  const float* w1     = (const float*)d_in[10];
  const float* b1     = (const float*)d_in[11];
  const float* w2     = (const float*)d_in[12];
  const float* b2     = (const float*)d_in[13];
  float* out = (float*)d_out;

  char* ws = (char*)d_ws;
  size_t off = 0;
  auto alloc = [&](size_t bytes) -> void* {
    void* p = ws + off;
    off = (off + bytes + 255) & ~(size_t)255;
    return p;
  };
  __bf16* h_bf   = (__bf16*)alloc((size_t)BT_ * E_ * 2);
  __bf16* q_bf   = (__bf16*)alloc((size_t)BT_ * E_ * 2);
  __bf16* k_bf   = (__bf16*)alloc((size_t)BT_ * E_ * 2);
  __bf16* vT_bf  = (__bf16*)alloc((size_t)BT_ * E_ * 2);
  __bf16* ao_bf  = (__bf16*)alloc((size_t)BT_ * E_ * 2);
  float*  x1     = (float*) alloc((size_t)BT_ * E_ * 4);
  __bf16* h2_bf  = (__bf16*)alloc((size_t)BT_ * E_ * 2);
  __bf16* ff1_bf = (__bf16*)alloc((size_t)BT_ * E_ * 2);
  __bf16* WqT    = (__bf16*)alloc((size_t)E_ * E_ * 2);
  __bf16* WkT    = (__bf16*)alloc((size_t)E_ * E_ * 2);
  __bf16* WvT    = (__bf16*)alloc((size_t)E_ * E_ * 2);
  __bf16* WpT    = (__bf16*)alloc((size_t)E_ * E_ * 2);
  __bf16* W1T    = (__bf16*)alloc((size_t)E_ * E_ * 2);
  __bf16* W2T    = (__bf16*)alloc((size_t)E_ * E_ * 2);

  const int nW = (E_ * E_) / 256;
  cvt_headmajor_T_kernel<<<nW, 256, 0, stream>>>(wq, WqT);
  cvt_headmajor_T_kernel<<<nW, 256, 0, stream>>>(wk, WkT);
  cvt_headmajor_T_kernel<<<nW, 256, 0, stream>>>(wv, WvT);
  cvt_plain_T_kernel<<<nW, 256, 0, stream>>>(w_proj, WpT);
  cvt_plain_T_kernel<<<nW, 256, 0, stream>>>(w1, W1T);
  cvt_plain_T_kernel<<<nW, 256, 0, stream>>>(w2, W2T);

  ln_kernel<<<BT_, 256, 0, stream>>>(x, ln1_g, ln1_b, h_bf);

  dim3 gGemm(E_ / BN, BT_ / BM);  // (8, 128)
  // Q, K -> bf16 [BT][E]
  gemm_bf16_kernel<false, false, false, false, true, false>
      <<<gGemm, 256, 0, stream>>>(h_bf, WqT, nullptr, nullptr, nullptr, q_bf, nullptr);
  gemm_bf16_kernel<false, false, false, false, true, false>
      <<<gGemm, 256, 0, stream>>>(h_bf, WkT, nullptr, nullptr, nullptr, k_bf, nullptr);
  // V -> transposed per-head [b][h][d][s]
  gemm_bf16_kernel<false, false, false, false, false, true>
      <<<gGemm, 256, 0, stream>>>(h_bf, WvT, nullptr, nullptr, nullptr, nullptr, vT_bf);

  attn_kernel<<<dim3(T_ / 64, B_ * H_), 128, 0, stream>>>(q_bf, k_bf, vT_bf, ao_bf);

  // x1 = x + attn @ Wp + b_proj
  gemm_bf16_kernel<true, true, false, true, false, false>
      <<<gGemm, 256, 0, stream>>>(ao_bf, WpT, b_proj, x, x1, nullptr, nullptr);
  ln_kernel<<<BT_, 256, 0, stream>>>(x1, ln2_g, ln2_b, h2_bf);
  // ff1 = relu(h2 @ W1 + b1)
  gemm_bf16_kernel<true, false, true, false, true, false>
      <<<gGemm, 256, 0, stream>>>(h2_bf, W1T, b1, nullptr, nullptr, ff1_bf, nullptr);
  // out = x1 + ff1 @ W2 + b2
  gemm_bf16_kernel<true, true, false, true, false, false>
      <<<gGemm, 256, 0, stream>>>(ff1_bf, W2T, b2, x1, out, nullptr, nullptr);
}
